// Transformer_56891136803289
// MI455X (gfx1250) — matrix operand order repaired and checked by reading, not compiled, not run on hardware
//
#include <hip/hip_runtime.h>
#include <hip/hip_bf16.h>
#include <math.h>

// ---------------- model constants ----------------
constexpr int Bb   = 2;
constexpr int Tt   = 1024;
constexpr int Cc   = 1024;
constexpr int Hh   = 16;
constexpr int DHh  = 64;       // head dim
constexpr int DFFf = 4096;
constexpr int Ll   = 8;

typedef __attribute__((ext_vector_type(16))) _Float16 v16h;
typedef __attribute__((ext_vector_type(8)))  _Float16 v8h;
typedef __attribute__((ext_vector_type(8)))  float    v8f;
typedef __attribute__((ext_vector_type(4)))  int      v4i;

// ---------------- gfx1250 async global->LDS path (guarded) ----------------
// Probe result (round 2 stderr): param0 of the async builtin is
// 'int __vector(4) __device__ *' i.e. AS(1) v4i*, so pass typed AS pointers.
typedef __attribute__((address_space(1))) v4i* as1_v4i_p;
typedef __attribute__((address_space(3))) v4i* as3_v4i_p;

#if defined(__AMDGCN__) && \
    __has_builtin(__builtin_amdgcn_global_load_async_to_lds_b128) && \
    __has_builtin(__builtin_amdgcn_s_wait_asynccnt)
#define HAVE_ASYNC 1
#define ASYNC_CP(gsrc, ldst)                                              \
    __builtin_amdgcn_global_load_async_to_lds_b128(                       \
        (as1_v4i_p)(gsrc), (as3_v4i_p)(ldst), 0, 0)
#define ASYNC_WAIT() __builtin_amdgcn_s_wait_asynccnt(0)
#else
#define HAVE_ASYNC 0
#define ASYNC_CP(gsrc, ldst) (*(v8h*)(ldst) = *(const v8h*)(gsrc))
#define ASYNC_WAIT() ((void)0)
#endif

// ---------------- LayerNorm (row = one block of 256) ----------------
__global__ __launch_bounds__(256)
void k_layernorm(const float* __restrict__ x, const float* __restrict__ sc,
                 const float* __restrict__ bi, float* __restrict__ oF32,
                 _Float16* __restrict__ oF16) {
    __shared__ float red[256];
    const int C = Cc;
    int row = blockIdx.x, tid = threadIdx.x;
    const float* xr = x + (size_t)row * C;
    float sum = 0.f, sq = 0.f;
    for (int i = tid; i < C; i += 256) { float v = xr[i]; sum += v; sq += v * v; }
    red[tid] = sum; __syncthreads();
    for (int o = 128; o > 0; o >>= 1) { if (tid < o) red[tid] += red[tid + o]; __syncthreads(); }
    float mean = red[0] * (1.f / C); __syncthreads();
    red[tid] = sq; __syncthreads();
    for (int o = 128; o > 0; o >>= 1) { if (tid < o) red[tid] += red[tid + o]; __syncthreads(); }
    float var = red[0] * (1.f / C) - mean * mean;
    float inv = rsqrtf(var + 1e-5f);
    for (int i = tid; i < C; i += 256) {
        float v = (xr[i] - mean) * inv * sc[i] + bi[i];
        if (oF32) oF32[(size_t)row * C + i] = v;
        if (oF16) oF16[(size_t)row * C + i] = (_Float16)v;
    }
}

// ---------------- tiled f32 -> f16 transpose: in[R][Cm] -> out[Cm][R] ----------------
__global__ __launch_bounds__(256)
void k_convert_transpose(const float* __restrict__ in, _Float16* __restrict__ out,
                         int R, int Cm) {
    __shared__ float tile[32][33];
    int bx = blockIdx.x, by = blockIdx.y;          // bx over Cm/32, by over R/32
    int tx = threadIdx.x & 31, ty = threadIdx.x >> 5;   // ty in 0..7
#pragma unroll
    for (int k = 0; k < 32; k += 8) {
        int r = by * 32 + ty + k, c = bx * 32 + tx;
        tile[ty + k][tx] = in[(size_t)r * Cm + c];
    }
    __syncthreads();
#pragma unroll
    for (int k = 0; k < 32; k += 8) {
        int r = bx * 32 + ty + k, c = by * 32 + tx;     // out row = in col
        out[(size_t)r * R + c] = (_Float16)tile[tx][ty + k];
    }
}

// ---------------- f16 transpose of V: (B,T,H,DH) -> (B,H,DH,T) ----------------
__global__ __launch_bounds__(256)
void k_transpose_v(const _Float16* __restrict__ v, _Float16* __restrict__ vt) {
    __shared__ _Float16 tile[32][33];
    int bx = blockIdx.x;                 // t tile (T/32)
    int byz = blockIdx.y;                // (b*H + h)*2 + dtile
    int dtile = byz & 1; int bh = byz >> 1;
    int b = bh / Hh, h = bh % Hh;
    int tx = threadIdx.x & 31, ty = threadIdx.x >> 5;
    const _Float16* src = v + ((size_t)b * Tt) * Cc + h * DHh + dtile * 32;
#pragma unroll
    for (int k = 0; k < 32; k += 8) {
        int t = bx * 32 + ty + k;
        tile[ty + k][tx] = src[(size_t)t * Cc + tx];    // [t_local][dh_local]
    }
    __syncthreads();
    _Float16* dst = vt + (((size_t)bh * DHh + dtile * 32)) * Tt;
#pragma unroll
    for (int k = 0; k < 32; k += 8) {
        int d = ty + k;
        int t = bx * 32 + tx;
        dst[(size_t)d * Tt + t] = tile[tx][d];
    }
}

// ---------------- WMMA GEMM: D = act(A[MxK] @ Wt^T + bias) (+resid) ----------------
// Wt is the PRE-TRANSPOSED weight, layout [N][K] row-major.
// Block = 256 threads = 8 waves. Macro tile 128x128, BK=32, double-buffered LDS.
// Wave (wm = wave&3, wn = wave>>2) computes rows [32*wm,+32) x cols [64*wn,+64).
template<int ACT, int RES>
__global__ __launch_bounds__(256)
void k_gemm_wmma(const _Float16* __restrict__ A, const _Float16* __restrict__ Wt,
                 const float* __restrict__ bias, const float* __restrict__ resid,
                 float* __restrict__ outF32, _Float16* __restrict__ outF16,
                 int M, int N, int K) {
    constexpr int BM = 128, BN = 128, BK = 32;
    __shared__ _Float16 As[2][BM][BK + 8];   // [m][k]
    __shared__ _Float16 Bs[2][BN][BK + 8];   // [n][k]  (Wt layout)
    const int tid = threadIdx.x, wave = tid >> 5, lane = tid & 31;
    const int hlf = lane >> 4;
    const int wm = wave & 3, wn = wave >> 2;
    const int m0 = blockIdx.y * BM, n0 = blockIdx.x * BN;
    v8f acc[2][4] = {};

    auto stageA = [&](int buf, int k0) {
#pragma unroll
        for (int it = 0; it < 2; ++it) {
            int idx = tid + it * 256;            // 512 chunks of 8 halves (128x32)
            int r = idx >> 2, c = (idx & 3) * 8;
            ASYNC_CP(&A[(size_t)(m0 + r) * K + k0 + c], &As[buf][r][c]);
        }
    };
    auto stageB = [&](int buf, int k0) {
#pragma unroll
        for (int it = 0; it < 2; ++it) {
            int idx = tid + it * 256;            // 512 chunks of 8 halves (128x32)
            int r = idx >> 2, c = (idx & 3) * 8;
            ASYNC_CP(&Wt[(size_t)(n0 + r) * K + k0 + c], &Bs[buf][r][c]);
        }
    };

    stageA(0, 0); stageB(0, 0);
    ASYNC_WAIT();
    __syncthreads();

    const int nsteps = K / BK;
    for (int s = 0; s < nsteps; ++s) {
        const int cur = s & 1, nxt = cur ^ 1;
        if (s + 1 < nsteps) { stageA(nxt, (s + 1) * BK); stageB(nxt, (s + 1) * BK); }
        // A fragments (16x32 f16): M = lane&15, K split by lane half
        v16h af[2];
#pragma unroll
        for (int mi = 0; mi < 2; ++mi) {
            int am = (wm << 5) + (mi << 4) + (lane & 15);
#pragma unroll
            for (int i = 0; i < 16; ++i) {
                int kk = i + (hlf << 4);
                af[mi][i] = As[cur][am][kk];
            }
        }
#pragma unroll
        for (int j = 0; j < 4; ++j) {
            // B fragment (32x16 f16): N = lane&15, K = 16*half + i  -> contiguous in Bs[n][k]
            v16h bf;
            int bn = (wn << 6) + (j << 4) + (lane & 15);
            int kb = hlf << 4;
#pragma unroll
            for (int i = 0; i < 16; ++i) bf[i] = Bs[cur][bn][kb + i];
#pragma unroll
            for (int mi = 0; mi < 2; ++mi)
                acc[mi][j] = __builtin_amdgcn_wmma_f32_16x16x32_f16(
                    false, af[mi], false, bf, (short)0, acc[mi][j], false, false);
        }
        ASYNC_WAIT();
        __syncthreads();
    }

    // epilogue: C/D layout -> row m = jj + 8*half, col n = lane&15
#pragma unroll
    for (int mi = 0; mi < 2; ++mi) {
#pragma unroll
        for (int j = 0; j < 4; ++j) {
            int n = n0 + (wn << 6) + (j << 4) + (lane & 15);
            float bv = bias[n];
#pragma unroll
            for (int jj = 0; jj < 8; ++jj) {
                int m = m0 + (wm << 5) + (mi << 4) + jj + (hlf << 3);
                float v = acc[mi][j][jj] + bv;
                if (ACT == 1) v = 0.5f * v * (1.0f + erff(v * 0.70710678118654752f));
                if (RES) v += resid[(size_t)m * N + n];
                if (outF32) outF32[(size_t)m * N + n] = v;
                if (outF16) outF16[(size_t)m * N + n] = (_Float16)v;
            }
        }
    }
}

// ---------------- RoPE on q,k (f16, layout B,T,H,DH) ----------------
__global__ __launch_bounds__(256)
void k_rope(_Float16* __restrict__ q, _Float16* __restrict__ k) {
    int idx = blockIdx.x * 256 + threadIdx.x;       // over B*T*H*(DH/2)
    const int PJ = DHh / 2;
    int total = Bb * Tt * Hh * PJ;
    if (idx >= total) return;
    int j = idx % PJ;
    int h = (idx / PJ) % Hh;
    int t = (idx / (PJ * Hh)) % Tt;
    int b = idx / (PJ * Hh * Tt);
    float inv = __expf(-(2.0f * (float)j / (float)DHh) * 9.210340371976184f);
    float ang = (float)t * inv;
    float c = __cosf(ang), s = __sinf(ang);
    size_t base = ((size_t)(b * Tt + t) * Cc) + h * DHh + 2 * j;
    float q0 = (float)q[base], q1 = (float)q[base + 1];
    q[base]     = (_Float16)(q0 * c - q1 * s);
    q[base + 1] = (_Float16)(q0 * s + q1 * c);
    float k0 = (float)k[base], k1 = (float)k[base + 1];
    k[base]     = (_Float16)(k0 * c - k1 * s);
    k[base + 1] = (_Float16)(k0 * s + k1 * c);
}

// ---------------- Flash attention (causal), WMMA, online softmax ----------------
// grid = (T/128, B*H); block = 256 threads = 8 waves; wave w owns q rows [16w,16w+16).
// v is pre-transposed: vt layout (B,H,DH,T).
__global__ __launch_bounds__(256)
void k_attn(const _Float16* __restrict__ q, const _Float16* __restrict__ k,
            const _Float16* __restrict__ vt, _Float16* __restrict__ y) {
    const int qt = blockIdx.x;
    const int bh = blockIdx.y;
    const int b = bh / Hh, h = bh % Hh;
    const int tid = threadIdx.x, wave = tid >> 5, lane = tid & 31;
    const int hlf = lane >> 4;

    __shared__ _Float16 Qs[128][64 + 8];       // [q_local][dh]
    __shared__ _Float16 Ks[64][64 + 8];        // [key_local][dh]
    __shared__ _Float16 Vts[64][64 + 8];       // [dh][key_local]
    __shared__ _Float16 Ps[8][16][64 + 8];     // per-wave P staging

    const _Float16* qb  = q  + ((size_t)b * Tt) * Cc + h * DHh;
    const _Float16* kb  = k  + ((size_t)b * Tt) * Cc + h * DHh;
    const _Float16* vtb = vt + ((size_t)bh * DHh) * Tt;

    // stage Q tile (128 x 64)
    for (int it = tid; it < 128 * 8; it += 256) {
        int r = it >> 3, c = (it & 7) * 8;
        ASYNC_CP(&qb[(size_t)(qt * 128 + r) * Cc + c], &Qs[r][c]);
    }

    float mrow[8], lrow[8];
    v8f o[4] = {};
#pragma unroll
    for (int jj = 0; jj < 8; ++jj) { mrow[jj] = -1e30f; lrow[jj] = 0.f; }

    const int ktmax = 2 * qt + 1;
    for (int kt = 0; kt <= ktmax; ++kt) {
        __syncthreads();   // protect Ks/Vts before overwrite
        for (int it = tid; it < 64 * 8; it += 256) {
            int r = it >> 3, c = (it & 7) * 8;
            ASYNC_CP(&kb[(size_t)(kt * 64 + r) * Cc + c], &Ks[r][c]);
            ASYNC_CP(&vtb[(size_t)r * Tt + kt * 64 + c], &Vts[r][c]);
        }
        ASYNC_WAIT();
        __syncthreads();

        // ---- S = Q_strip(16xDH) @ K^T(DHx64), DH contraction = 2 x 32 ----
        v8f sacc[4] = {};
#pragma unroll
        for (int ks = 0; ks < 2; ++ks) {
            v16h af;
            int am = (wave << 4) + (lane & 15);
#pragma unroll
            for (int i = 0; i < 16; ++i) {
                int kk = ks * 32 + i + (hlf << 4);
                af[i] = Qs[am][kk];
            }
#pragma unroll
            for (int j = 0; j < 4; ++j) {
                v16h bf;
                int n = (j << 4) + (lane & 15);      // key within tile
#pragma unroll
                for (int i = 0; i < 16; ++i) bf[i] = Ks[n][ks * 32 + (hlf << 4) + i];
                sacc[j] = __builtin_amdgcn_wmma_f32_16x16x32_f16(
                    false, af, false, bf, (short)0, sacc[j], false, false);
            }
        }

        // ---- scale, causal mask, tile row-max ----
        const float scale = 0.125f;                 // 1/sqrt(64)
        float tmax[8];
#pragma unroll
        for (int jj = 0; jj < 8; ++jj) tmax[jj] = -1e30f;
#pragma unroll
        for (int j = 0; j < 4; ++j) {
            int kg = kt * 64 + (j << 4) + (lane & 15);
#pragma unroll
            for (int jj = 0; jj < 8; ++jj) {
                int qg = qt * 128 + (wave << 4) + jj + (hlf << 3);
                float sv = sacc[j][jj] * scale;
                if (kg > qg) sv = -1e30f;
                sacc[j][jj] = sv;
                tmax[jj] = fmaxf(tmax[jj], sv);
            }
        }
#pragma unroll
        for (int jj = 0; jj < 8; ++jj) {
            float mv = tmax[jj];
            for (int off = 1; off < 16; off <<= 1)
                mv = fmaxf(mv, __shfl_xor(mv, off, 32));
            tmax[jj] = mv;
        }

        float rescale[8];
#pragma unroll
        for (int jj = 0; jj < 8; ++jj) {
            float mnew = fmaxf(mrow[jj], tmax[jj]);
            rescale[jj] = __expf(mrow[jj] - mnew);
            mrow[jj] = mnew;
        }

        // ---- P = exp(S - m); stage to LDS in natural (row, col) order ----
        float psum[8];
#pragma unroll
        for (int jj = 0; jj < 8; ++jj) psum[jj] = 0.f;
#pragma unroll
        for (int j = 0; j < 4; ++j) {
#pragma unroll
            for (int jj = 0; jj < 8; ++jj) {
                float p = __expf(sacc[j][jj] - mrow[jj]);
                psum[jj] += p;
                Ps[wave][jj + (hlf << 3)][(j << 4) + (lane & 15)] = (_Float16)p;
            }
        }
#pragma unroll
        for (int jj = 0; jj < 8; ++jj) {
            float sv = psum[jj];
            for (int off = 1; off < 16; off <<= 1) sv += __shfl_xor(sv, off, 32);
            lrow[jj] = lrow[jj] * rescale[jj] + sv;
        }
#pragma unroll
        for (int d = 0; d < 4; ++d)
#pragma unroll
            for (int jj = 0; jj < 8; ++jj) o[d][jj] *= rescale[jj];

        __syncthreads();   // Ps visible for A-frag re-gather

        // ---- O += P(16x64) @ V(64xDH), key contraction = 2 x 32 ----
#pragma unroll
        for (int ks = 0; ks < 2; ++ks) {
            v16h af;
            int am = lane & 15;
#pragma unroll
            for (int i = 0; i < 16; ++i) {
                int kk = ks * 32 + i + (hlf << 4);
                af[i] = Ps[wave][am][kk];
            }
#pragma unroll
            for (int d = 0; d < 4; ++d) {
                v16h bf;
                int n = (d << 4) + (lane & 15);      // dh column; Vts[dh][key] contiguous
#pragma unroll
                for (int i = 0; i < 16; ++i) bf[i] = Vts[n][ks * 32 + (hlf << 4) + i];
                o[d] = __builtin_amdgcn_wmma_f32_16x16x32_f16(
                    false, af, false, bf, (short)0, o[d], false, false);
            }
        }
    }

    // ---- normalize + store y (B,T,H,DH) as f16 ----
    _Float16* yb = y + ((size_t)b * Tt) * Cc + h * DHh;
#pragma unroll
    for (int d = 0; d < 4; ++d) {
        int dd = (d << 4) + (lane & 15);
#pragma unroll
        for (int jj = 0; jj < 8; ++jj) {
            int qg = qt * 128 + (wave << 4) + jj + (hlf << 3);
            yb[(size_t)qg * Cc + dd] = (_Float16)(o[d][jj] / lrow[jj]);
        }
    }
}

// ---------------- host-side launch ----------------
extern "C" void kernel_launch(void* const* d_in, const int* in_sizes, int n_in,
                              void* d_out, int out_size, void* d_ws, size_t ws_size,
                              hipStream_t stream) {
    const float* x    = (const float*)d_in[0];
    const float* Wq   = (const float*)d_in[1];  const float* bq  = (const float*)d_in[2];
    const float* Wk   = (const float*)d_in[3];  const float* bk  = (const float*)d_in[4];
    const float* Wv   = (const float*)d_in[5];  const float* bv  = (const float*)d_in[6];
    const float* Wo   = (const float*)d_in[7];  const float* bo  = (const float*)d_in[8];
    const float* ln1s = (const float*)d_in[9];  const float* ln1b = (const float*)d_in[10];
    const float* ln2s = (const float*)d_in[11]; const float* ln2b = (const float*)d_in[12];
    const float* W1   = (const float*)d_in[13]; const float* b1  = (const float*)d_in[14];
    const float* W2   = (const float*)d_in[15]; const float* b2  = (const float*)d_in[16];
    const float* lnfs = (const float*)d_in[17]; const float* lnfb = (const float*)d_in[18];

    const int BT = Bb * Tt;                       // 2048 rows
    char* wsp = (char*)d_ws;
    auto carve = [&](size_t bytes) -> void* {
        void* p = (void*)wsp;
        wsp += (bytes + 255) & ~(size_t)255;
        return p;
    };
    float*     hbuf = (float*)    carve((size_t)BT * Cc * sizeof(float));
    _Float16*  a16  = (_Float16*) carve((size_t)BT * Cc * 2);
    _Float16*  q16  = (_Float16*) carve((size_t)BT * Cc * 2);
    _Float16*  k16  = (_Float16*) carve((size_t)BT * Cc * 2);
    _Float16*  v16  = (_Float16*) carve((size_t)BT * Cc * 2);
    _Float16*  vt16 = (_Float16*) carve((size_t)BT * Cc * 2);
    _Float16*  y16  = (_Float16*) carve((size_t)BT * Cc * 2);
    _Float16*  m16  = (_Float16*) carve((size_t)BT * DFFf * 2);
    _Float16*  wqt  = (_Float16*) carve((size_t)Cc * Cc * 2);
    _Float16*  wkt  = (_Float16*) carve((size_t)Cc * Cc * 2);
    _Float16*  wvt  = (_Float16*) carve((size_t)Cc * Cc * 2);
    _Float16*  wot  = (_Float16*) carve((size_t)Cc * Cc * 2);
    _Float16*  w1t  = (_Float16*) carve((size_t)Cc * DFFf * 2);
    _Float16*  w2t  = (_Float16*) carve((size_t)DFFf * Cc * 2);

    // h = x  (residual stream, fp32)
    (void)hipMemcpyAsync(hbuf, x, (size_t)BT * Cc * sizeof(float),
                         hipMemcpyDeviceToDevice, stream);

    const int CC = Cc * Cc, CF = Cc * DFFf;
    dim3 tr_sq(Cc / 32, Cc / 32);                 // transpose grids
    dim3 tr_1(DFFf / 32, Cc / 32);                // W1: (K=Cc) x (N=DFFf)
    dim3 tr_2(Cc / 32, DFFf / 32);                // W2: (K=DFFf) x (N=Cc)
    dim3 gemm_g(Cc / 128, BT / 128);              // N=1024 GEMMs: 8 x 16
    dim3 gemm_g_ff(DFFf / 128, BT / 128);         // fc1: 32 x 16
    dim3 attn_g(Tt / 128, Bb * Hh);               // 8 x 32
    dim3 trv_g(Tt / 32, Bb * Hh * (DHh / 32));    // 32 x 64

    for (int l = 0; l < Ll; ++l) {
        // per-layer weight convert+transpose to f16 [N][K]
        k_convert_transpose<<<tr_sq, 256, 0, stream>>>(Wq + (size_t)l * CC, wqt, Cc, Cc);
        k_convert_transpose<<<tr_sq, 256, 0, stream>>>(Wk + (size_t)l * CC, wkt, Cc, Cc);
        k_convert_transpose<<<tr_sq, 256, 0, stream>>>(Wv + (size_t)l * CC, wvt, Cc, Cc);
        k_convert_transpose<<<tr_sq, 256, 0, stream>>>(Wo + (size_t)l * CC, wot, Cc, Cc);
        k_convert_transpose<<<tr_1, 256, 0, stream>>>(W1 + (size_t)l * CF, w1t, Cc, DFFf);
        k_convert_transpose<<<tr_2, 256, 0, stream>>>(W2 + (size_t)l * CF, w2t, DFFf, Cc);

        // ln1 -> a16 (f16)
        k_layernorm<<<BT, 256, 0, stream>>>(hbuf, ln1s + (size_t)l * Cc, ln1b + (size_t)l * Cc,
                                            nullptr, a16);
        // QKV projections (f16 out)
        k_gemm_wmma<0, 0><<<gemm_g, 256, 0, stream>>>(a16, wqt, bq + (size_t)l * Cc,
                                                      nullptr, nullptr, q16, BT, Cc, Cc);
        k_gemm_wmma<0, 0><<<gemm_g, 256, 0, stream>>>(a16, wkt, bk + (size_t)l * Cc,
                                                      nullptr, nullptr, k16, BT, Cc, Cc);
        k_gemm_wmma<0, 0><<<gemm_g, 256, 0, stream>>>(a16, wvt, bv + (size_t)l * Cc,
                                                      nullptr, nullptr, v16, BT, Cc, Cc);
        // RoPE on q,k
        {
            int total = Bb * Tt * Hh * (DHh / 2);
            k_rope<<<(total + 255) / 256, 256, 0, stream>>>(q16, k16);
        }
        // V -> (B,H,DH,T)
        k_transpose_v<<<trv_g, 256, 0, stream>>>(v16, vt16);
        // flash attention -> y16
        k_attn<<<attn_g, 256, 0, stream>>>(q16, k16, vt16, y16);
        // out projection + residual: h = h + y @ Wo + bo
        k_gemm_wmma<0, 1><<<gemm_g, 256, 0, stream>>>(y16, wot, bo + (size_t)l * Cc,
                                                      hbuf, hbuf, nullptr, BT, Cc, Cc);
        // ln2 -> a16
        k_layernorm<<<BT, 256, 0, stream>>>(hbuf, ln2s + (size_t)l * Cc, ln2b + (size_t)l * Cc,
                                            nullptr, a16);
        // fc1 + exact GELU -> m16 (f16)
        k_gemm_wmma<1, 0><<<gemm_g_ff, 256, 0, stream>>>(a16, w1t, b1 + (size_t)l * DFFf,
                                                         nullptr, nullptr, m16, BT, DFFf, Cc);
        // fc2 + residual: h = h + m @ W2 + b2
        k_gemm_wmma<0, 1><<<gemm_g, 256, 0, stream>>>(m16, w2t, b2 + (size_t)l * Cc,
                                                      hbuf, hbuf, nullptr, BT, Cc, DFFf);
    }
    // final LayerNorm -> d_out (fp32)
    k_layernorm<<<BT, 256, 0, stream>>>(hbuf, lnfs, lnfb, (float*)d_out, nullptr);
}
